// MaskedWSABlock_63513976373836
// MI455X (gfx1250) — matrix operand order repaired and checked by reading, not compiled, not run on hardware
//
#include <hip/hip_runtime.h>

// ---------------------------------------------------------------------------
// Swin-style masked windowed shifted attention block for MI455X (gfx1250).
// GEMMs: v_wmma_f32_16x16x32_bf16 (wave32), 32x32 wave tiles, fp32 accum.
// Tile staging: Tensor Data Mover (tensor_load_to_lds) with double-buffered
// LDS, tracked by TENSORcnt.  Set USE_TDM 0 to fall back to vector loads.
// ---------------------------------------------------------------------------

#define USE_TDM 1

typedef __attribute__((ext_vector_type(16))) __bf16 v16bf;
typedef __attribute__((ext_vector_type(8)))  float  v8f;
typedef __attribute__((ext_vector_type(4)))  unsigned int v4u;
typedef __attribute__((ext_vector_type(8)))  int    v8i;
typedef __attribute__((ext_vector_type(4)))  int    v4i;

#define B_BATCH 32
#define HH 56
#define CC 256
#define WIN 7
#define SHIFT 3
#define HEADS 8
#define NTOK 49
#define NW 64
#define BWIN 2048
#define LTOK 3136
#define MTOK 100352
#define DHEAD 32
#define QSCALE 0.17677669529663687f   // 32^-0.5

__device__ __forceinline__ unsigned short f2bf(float f) {
    unsigned int u = __builtin_bit_cast(unsigned int, f);
    unsigned int r = u + 0x7FFFu + ((u >> 16) & 1u);   // RNE
    return (unsigned short)(r >> 16);
}

union Frag32B {            // one WMMA 16-bit operand = 32 bytes
    uint4  q[2];
    v16bf  bf;
};
union Pack8 {              // 8 bf16 = 16 bytes
    unsigned short us[8];
    uint4 q;
};

__device__ __forceinline__ v8f wmma_bf16(v16bf a, v16bf b, v8f c) {
    return __builtin_amdgcn_wmma_f32_16x16x32_bf16(
        false, a, false, b, (short)0, c, false, false);
}

#if USE_TDM
// 2-D bf16 tile DMA: global (row-major, stride_elems per row) -> LDS
// (contiguous tile_w-element rows).  D# packing per CDNA5 ISA ch.8.
// amdgpu-toolchain (clang-23) 6-arg builtin form.
__device__ __forceinline__ void tdm_load_tile_bf16(
    unsigned lds_addr, const unsigned short* gsrc,
    unsigned tile_w, unsigned tile_h, unsigned stride_elems)
{
    unsigned long long ga = (unsigned long long)(size_t)gsrc;
    v4u g0 = { 1u,                                   // count=1, user descriptor
               lds_addr,                             // lds_addr[31:0]
               (unsigned)ga,                         // global_addr lo
               (((unsigned)(ga >> 32)) & 0x01FFFFFFu) | 0x80000000u }; // hi + type=2
    unsigned td0 = stride_elems;       // tensor_dim0 (>= tile_w, no OOB)
    unsigned td1 = 0x7FFFFFFFu;        // tensor_dim1 (large, no OOB)
    v8i g1;
    g1[0] = (int)(1u << 16);                                       // data_size=2B
    g1[1] = (int)((td0 & 0xFFFFu) << 16);                          // dim0 lo16
    g1[2] = (int)(((td0 >> 16) & 0xFFFFu) | ((td1 & 0xFFFFu) << 16));
    g1[3] = (int)(((td1 >> 16) & 0xFFFFu) | (tile_w << 16));       // tile_dim0
    g1[4] = (int)(tile_h & 0xFFFFu);                               // tile_dim1
    g1[5] = (int)stride_elems;                                     // dim0_stride lo32
    g1[6] = 0;
    g1[7] = 0;
    v4i z4 = {0, 0, 0, 0};
    v8i z8 = {0, 0, 0, 0, 0, 0, 0, 0};
    __builtin_amdgcn_tensor_load_to_lds(g0, g1, z4, z4, z8, 0);
}
#endif

// ---------------------------------------------------------------------------
// fp32 -> bf16 weight conversion (one-time, ~1.5 MB total)
// ---------------------------------------------------------------------------
__global__ __launch_bounds__(256) void cvt_bf16_kernel(
    const float* __restrict__ src, unsigned short* __restrict__ dst, int n)
{
    int idx = (blockIdx.x * 256 + threadIdx.x) * 4;
    if (idx >= n) return;
    float4 f = *(const float4*)(src + idx);
    unsigned short o0 = f2bf(f.x), o1 = f2bf(f.y), o2 = f2bf(f.z), o3 = f2bf(f.w);
    uint2 pk = { (unsigned)o0 | ((unsigned)o1 << 16),
                 (unsigned)o2 | ((unsigned)o3 << 16) };
    *(uint2*)(dst + idx) = pk;
}

// ---------------------------------------------------------------------------
// LayerNorm, one wave per token. windowed==1: read token-major x, write to
// shifted + window-partitioned row layout (B_, n) in bf16.
// ---------------------------------------------------------------------------
__global__ __launch_bounds__(256) void ln_kernel(
    const float* __restrict__ x, const float* __restrict__ g,
    const float* __restrict__ bta, unsigned short* __restrict__ out,
    int windowed)
{
    int gw   = (blockIdx.x * blockDim.x + threadIdx.x) >> 5;
    int lane = threadIdx.x & 31;
    if (gw >= MTOK) return;

    size_t srow;
    if (windowed) {
        int b_ = gw / NTOK, n = gw % NTOK;
        int bb = b_ >> 6, wi = b_ & 63;
        int i = n / WIN, j = n % WIN;
        int hs = (wi >> 3) * WIN + i, wsv = (wi & 7) * WIN + j;
        int hh = (hs + SHIFT) % HH, ww = (wsv + SHIFT) % HH;
        srow = (size_t)bb * LTOK + hh * HH + ww;
    } else {
        srow = (size_t)gw;
    }
    const float4* r4 = (const float4*)(x + srow * CC);
    float4 va = r4[lane * 2];
    float4 vb = r4[lane * 2 + 1];
    float v[8] = {va.x, va.y, va.z, va.w, vb.x, vb.y, vb.z, vb.w};

    float s = 0.f, sq = 0.f;
    #pragma unroll
    for (int e = 0; e < 8; e++) { s += v[e]; sq += v[e] * v[e]; }
    #pragma unroll
    for (int off = 16; off; off >>= 1) {
        s  += __shfl_xor(s,  off, 32);
        sq += __shfl_xor(sq, off, 32);
    }
    float m   = s * (1.f / CC);
    float var = sq * (1.f / CC) - m * m;
    float inv = rsqrtf(var + 1e-5f);

    const float4* g4 = (const float4*)g;
    const float4* b4 = (const float4*)bta;
    float4 g0 = g4[lane * 2], g1 = g4[lane * 2 + 1];
    float4 t0 = b4[lane * 2], t1 = b4[lane * 2 + 1];
    float gg[8] = {g0.x, g0.y, g0.z, g0.w, g1.x, g1.y, g1.z, g1.w};
    float tt[8] = {t0.x, t0.y, t0.z, t0.w, t1.x, t1.y, t1.z, t1.w};

    Pack8 o;
    #pragma unroll
    for (int e = 0; e < 8; e++) o.us[e] = f2bf((v[e] - m) * inv * gg[e] + tt[e]);
    *(uint4*)(out + (size_t)gw * CC + lane * 8) = o.q;
}

// ---------------------------------------------------------------------------
// Tiled bf16 WMMA GEMM:  out[m,n] = sum_k A[m,k] * Wb[n,k]  (+ bias, epilogue)
// Block tile 128x64, 8 waves (4 row x 2 col), wave tile 32x32 = 4 WMMA/K-step.
// A tile [m][k] row-major; B tile [n][k] (K-contiguous).  All frags 2 x b128.
// TDM double-buffers both tiles; wave 0 drives the DMA + TENSORcnt.
// modes: 0 = QKV split/scale, 1 = proj + window-reverse + residual,
//        2 = fc1 + GELU, 3 = fc2 + residual
// ---------------------------------------------------------------------------
__global__ __launch_bounds__(256) void gemm_bf16_kernel(
    const unsigned short* __restrict__ A, const unsigned short* __restrict__ Wb,
    const float* __restrict__ bias, int M, int Nout, int K, int mode,
    unsigned short* __restrict__ ou0, unsigned short* __restrict__ ou1,
    unsigned short* __restrict__ ou2, float* __restrict__ of,
    const float* __restrict__ res)
{
    __shared__ __align__(16) unsigned short As[2][128][32];   // [buf][m][k]
    __shared__ __align__(16) unsigned short Bt[2][64][32];    // [buf][n][k]

    const int mBase = blockIdx.x * 128;
    const int nBase = blockIdx.y * 64;
    const int tid   = threadIdx.x;
    const int lane  = tid & 31;
    const int wid   = tid >> 5;
    const int wrow  = wid & 3;    // 32-row group
    const int wcol  = wid >> 2;   // 32-col group

    const int rbase = wrow * 32 + (lane & 15);
    const int khalf = (lane >> 4) * 16;
    const int kb    = (lane >> 4) * 16;
    const int n0    = wcol * 32 + (lane & 15);

    v8f acc[2][2];
    #pragma unroll
    for (int a = 0; a < 2; a++)
        #pragma unroll
        for (int b = 0; b < 2; b++)
            acc[a][b] = v8f{0.f,0.f,0.f,0.f,0.f,0.f,0.f,0.f};

    const int nsteps = K >> 5;

#if USE_TDM
    if (wid == 0) {
        tdm_load_tile_bf16((unsigned)(size_t)(void*)&As[0][0][0],
                           A + (size_t)mBase * K, 32, 128, (unsigned)K);
        tdm_load_tile_bf16((unsigned)(size_t)(void*)&Bt[0][0][0],
                           Wb + (size_t)nBase * K, 32, 64, (unsigned)K);
    }
#endif

    for (int s = 0; s < nsteps; s++) {
#if USE_TDM
        const int cur = s & 1;
        if (wid == 0) {
            if (s + 1 < nsteps) {
                int k1 = (s + 1) << 5;
                int nxt = cur ^ 1;
                tdm_load_tile_bf16((unsigned)(size_t)(void*)&As[nxt][0][0],
                                   A + (size_t)mBase * K + k1, 32, 128, (unsigned)K);
                tdm_load_tile_bf16((unsigned)(size_t)(void*)&Bt[nxt][0][0],
                                   Wb + (size_t)nBase * K + k1, 32, 64, (unsigned)K);
                __builtin_amdgcn_s_wait_tensorcnt(2);   // tile s landed
            } else {
                __builtin_amdgcn_s_wait_tensorcnt(0);
            }
        }
        __syncthreads();
#else
        const int cur = 0;
        const int k0 = s << 5;
        {   // stage A: 128x32, two b128 per thread
            int linear = tid * 16;
            int r = linear >> 5, kk = linear & 31;
            const unsigned short* src = A + (size_t)(mBase + r) * K + k0 + kk;
            *(uint4*)&As[0][r][kk]     = *(const uint4*)src;
            *(uint4*)&As[0][r][kk + 8] = *(const uint4*)(src + 8);
        }
        {   // stage B: 64x32, one b128 per thread
            int linear = tid * 8;
            int nn = linear >> 5, kk = linear & 31;
            *(uint4*)&Bt[0][nn][kk] =
                *(const uint4*)(Wb + (size_t)(nBase + nn) * K + k0 + kk);
        }
        __syncthreads();
#endif
        Frag32B fa0, fa1, fb0, fb1;
        fa0.q[0] = *(const uint4*)&As[cur][rbase][khalf];
        fa0.q[1] = *(const uint4*)&As[cur][rbase][khalf + 8];
        fa1.q[0] = *(const uint4*)&As[cur][rbase + 16][khalf];
        fa1.q[1] = *(const uint4*)&As[cur][rbase + 16][khalf + 8];
        fb0.q[0] = *(const uint4*)&Bt[cur][n0][kb];
        fb0.q[1] = *(const uint4*)&Bt[cur][n0][kb + 8];
        fb1.q[0] = *(const uint4*)&Bt[cur][n0 + 16][kb];
        fb1.q[1] = *(const uint4*)&Bt[cur][n0 + 16][kb + 8];

        acc[0][0] = wmma_bf16(fa0.bf, fb0.bf, acc[0][0]);
        acc[0][1] = wmma_bf16(fa0.bf, fb1.bf, acc[0][1]);
        acc[1][0] = wmma_bf16(fa1.bf, fb0.bf, acc[1][0]);
        acc[1][1] = wmma_bf16(fa1.bf, fb1.bf, acc[1][1]);
        __syncthreads();
    }

    // ---- epilogue ----
    #pragma unroll
    for (int rt = 0; rt < 2; rt++) {
        #pragma unroll
        for (int ct = 0; ct < 2; ct++) {
            #pragma unroll
            for (int r = 0; r < 8; r++) {
                int row = mBase + wrow * 32 + rt * 16 + ((lane < 16) ? r : 8 + r);
                int col = nBase + wcol * 32 + ct * 16 + (lane & 15);
                float val = acc[rt][ct][r] + bias[col];

                if (mode == 0) {                       // QKV split
                    int which = col >> 8;
                    int head  = (col >> 5) & 7;
                    int dch   = col & 31;
                    int b_ = row / NTOK, n = row % NTOK;
                    size_t di = ((size_t)(b_ * HEADS + head) * NTOK + n) * DHEAD + dch;
                    if      (which == 0) ou0[di] = f2bf(val * QSCALE);
                    else if (which == 1) ou1[di] = f2bf(val);
                    else                 ou2[di] = f2bf(val);
                } else if (mode == 1) {                // proj + reverse + residual
                    int b_ = row / NTOK, n = row % NTOK;
                    int bb = b_ >> 6, wi = b_ & 63;
                    int i = n / WIN, j = n % WIN;
                    int hs = (wi >> 3) * WIN + i, wsv = (wi & 7) * WIN + j;
                    int hh = (hs + SHIFT) % HH, ww = (wsv + SHIFT) % HH;
                    size_t di = ((size_t)bb * LTOK + hh * HH + ww) * CC + col;
                    of[di] = val + res[di];
                } else if (mode == 2) {                // fc1 + exact GELU
                    float gl = 0.5f * val * (1.f + erff(val * 0.70710678118654752f));
                    ou0[(size_t)row * Nout + col] = f2bf(gl);
                } else {                               // fc2 + residual
                    size_t di = (size_t)row * CC + col;
                    of[di] = val + res[di];
                }
            }
        }
    }
}

// ---------------------------------------------------------------------------
// Attention per (window, head). 128 threads = 4 waves; wave w owns row tile w.
// S (fp32, stride 68) holds scores with rpb + shift mask fused. Softmax writes
// combined P = p + p_fg - p_bg as bf16 into Pb; V staged transposed (Vt[d][m]).
// All fragments are 2 x b128 loads.
// ---------------------------------------------------------------------------
__global__ __launch_bounds__(128) void attn_kernel(
    const unsigned short* __restrict__ q, const unsigned short* __restrict__ k,
    const unsigned short* __restrict__ v, const float* __restrict__ rpb,
    const float* __restrict__ amask, const float* __restrict__ salfg,
    const float* __restrict__ salbg, unsigned short* __restrict__ outp)
{
    __shared__ __align__(16) float S[64][68];
    __shared__ __align__(16) unsigned short Pb[64][64];
    __shared__ __align__(16) unsigned short Vt[32][64];   // [d][m]
    __shared__ float fgm[64], bgm[64];

    const int bh   = blockIdx.x;
    const int b_   = bh >> 3;
    const int h    = bh & 7;
    const int wi   = b_ & 63;
    const int tid  = threadIdx.x;
    const int w    = tid >> 5;
    const int lane = tid & 31;

    const unsigned short* qb = q + (size_t)bh * NTOK * DHEAD;
    const unsigned short* kp = k + (size_t)bh * NTOK * DHEAD;
    const unsigned short* vp = v + (size_t)bh * NTOK * DHEAD;

    if (tid < 64) {
        fgm[tid] = (tid < NTOK) ? salfg[(size_t)b_ * NTOK * NTOK + tid] : 0.f;
        bgm[tid] = (tid < NTOK) ? salbg[(size_t)b_ * NTOK * NTOK + tid] : 0.f;
    }
    for (int idx = tid; idx < 64 * DHEAD; idx += 128) {
        int mm = idx >> 5;
        int dd = idx & 31;
        Vt[dd][mm] = (mm < NTOK) ? vp[(size_t)mm * DHEAD + dd] : (unsigned short)0;
    }

    const int khalf = (lane >> 4) * 16;
    Frag32B fa;
    {
        int n = w * 16 + (lane & 15);
        if (n < NTOK) {
            const unsigned short* qr = qb + (size_t)n * DHEAD;
            fa.q[0] = *(const uint4*)(qr + khalf);
            fa.q[1] = *(const uint4*)(qr + khalf + 8);
        } else {
            fa.q[0] = uint4{0u,0u,0u,0u};
            fa.q[1] = uint4{0u,0u,0u,0u};
        }
    }

    const int kb2 = (lane >> 4) * 16;
    #pragma unroll
    for (int ct = 0; ct < 4; ct++) {
        int mcol = ct * 16 + (lane & 15);
        Frag32B fb;
        if (mcol < NTOK) {
            const unsigned short* kr = kp + (size_t)mcol * DHEAD;
            fb.q[0] = *(const uint4*)(kr + kb2);
            fb.q[1] = *(const uint4*)(kr + kb2 + 8);
        } else {
            fb.q[0] = uint4{0u,0u,0u,0u};
            fb.q[1] = uint4{0u,0u,0u,0u};
        }
        v8f c = {0.f,0.f,0.f,0.f,0.f,0.f,0.f,0.f};
        c = wmma_bf16(fa.bf, fb.bf, c);

        #pragma unroll
        for (int r = 0; r < 8; r++) {
            int nn = w * 16 + ((lane < 16) ? r : 8 + r);
            int mm = ct * 16 + (lane & 15);
            float sv = -1e30f;
            if (nn < NTOK && mm < NTOK) {
                int i1 = nn / WIN, j1 = nn % WIN, i2 = mm / WIN, j2 = mm % WIN;
                int ridx = (i1 - i2 + WIN - 1) * (2 * WIN - 1) + (j1 - j2 + WIN - 1);
                sv = c[r] + rpb[ridx * HEADS + h]
                          + amask[((size_t)wi * NTOK + nn) * NTOK + mm];
            }
            S[nn][mm] = sv;
        }
    }
    __syncthreads();

    if (tid < 64) {
        int nn = tid;
        if (nn < NTOK) {
            float m1 = -1e30f, m2 = -1e30f, m3 = -1e30f;
            for (int mm = 0; mm < NTOK; mm++) {
                float s0 = S[nn][mm];
                m1 = fmaxf(m1, s0);
                m2 = fmaxf(m2, s0 + fgm[mm]);
                m3 = fmaxf(m3, s0 + bgm[mm]);
            }
            float s1 = 0.f, s2 = 0.f, s3 = 0.f;
            for (int mm = 0; mm < NTOK; mm++) {
                float s0 = S[nn][mm];
                s1 += __expf(s0 - m1);
                s2 += __expf(s0 + fgm[mm] - m2);
                s3 += __expf(s0 + bgm[mm] - m3);
            }
            float r1 = 1.f / s1, r2 = 1.f / s2, r3 = 1.f / s3;
            for (int mm = 0; mm < 64; mm++) {
                float p = 0.f;
                if (mm < NTOK) {
                    float s0 = S[nn][mm];
                    p = __expf(s0 - m1) * r1 + __expf(s0 + fgm[mm] - m2) * r2
                      - __expf(s0 + bgm[mm] - m3) * r3;
                }
                Pb[nn][mm] = f2bf(p);
            }
        } else {
            for (int mm = 0; mm < 64; mm++) Pb[nn][mm] = (unsigned short)0;
        }
    }
    __syncthreads();

    v8f o0 = {0.f,0.f,0.f,0.f,0.f,0.f,0.f,0.f};
    v8f o1 = {0.f,0.f,0.f,0.f,0.f,0.f,0.f,0.f};
    const int rl  = w * 16 + (lane & 15);
    const int kb3 = (lane >> 4) * 16;
    #pragma unroll
    for (int ks = 0; ks < 2; ks++) {
        Frag32B pa;
        pa.q[0] = *(const uint4*)&Pb[rl][ks * 32 + khalf];
        pa.q[1] = *(const uint4*)&Pb[rl][ks * 32 + khalf + 8];
        #pragma unroll
        for (int ct = 0; ct < 2; ct++) {
            int dch = ct * 16 + (lane & 15);
            Frag32B fv;
            fv.q[0] = *(const uint4*)&Vt[dch][ks * 32 + kb3];
            fv.q[1] = *(const uint4*)&Vt[dch][ks * 32 + kb3 + 8];
            if (ct == 0) o0 = wmma_bf16(pa.bf, fv.bf, o0);
            else         o1 = wmma_bf16(pa.bf, fv.bf, o1);
        }
    }

    v8f os[2] = {o0, o1};
    #pragma unroll
    for (int ct = 0; ct < 2; ct++) {
        #pragma unroll
        for (int r = 0; r < 8; r++) {
            int nn = w * 16 + ((lane < 16) ? r : 8 + r);
            if (nn < NTOK) {
                int dch = ct * 16 + (lane & 15);
                outp[((size_t)b_ * NTOK + nn) * CC + h * DHEAD + dch] = f2bf(os[ct][r]);
            }
        }
    }
}

// ---------------------------------------------------------------------------
extern "C" void kernel_launch(void* const* d_in, const int* in_sizes, int n_in,
                              void* d_out, int out_size, void* d_ws, size_t ws_size,
                              hipStream_t stream)
{
    const float* x      = (const float*)d_in[0];
    const float* qkv_w  = (const float*)d_in[1];
    const float* qkv_b  = (const float*)d_in[2];
    const float* rpb    = (const float*)d_in[3];
    const float* proj_w = (const float*)d_in[4];
    const float* proj_b = (const float*)d_in[5];
    const float* n1g    = (const float*)d_in[6];
    const float* n1b    = (const float*)d_in[7];
    const float* n2g    = (const float*)d_in[8];
    const float* n2b    = (const float*)d_in[9];
    const float* fc1w   = (const float*)d_in[10];
    const float* fc1b   = (const float*)d_in[11];
    const float* fc2w   = (const float*)d_in[12];
    const float* fc2b   = (const float*)d_in[13];
    const float* amask  = (const float*)d_in[14];
    const float* salfg  = (const float*)d_in[15];
    const float* salbg  = (const float*)d_in[16];
    float* out = (float*)d_out;

    char* ws = (char*)d_ws;
    const size_t SZ = (size_t)MTOK * CC * 2;                 // one bf16 token buffer
    unsigned short* xw   = (unsigned short*)(ws);            // xw/attn_out/ln2 (reused)
    unsigned short* qb   = (unsigned short*)(ws + SZ);
    unsigned short* kb   = (unsigned short*)(ws + 2 * SZ);
    unsigned short* vb   = (unsigned short*)(ws + 3 * SZ);
    float*          x2   = (float*)(ws + 4 * SZ);            // MTOK*CC fp32
    unsigned short* fc1o = (unsigned short*)(ws + 4 * SZ + (size_t)MTOK * CC * 4);
    unsigned short* wbf  = (unsigned short*)(ws + 4 * SZ + (size_t)MTOK * CC * 4
                                                + (size_t)MTOK * 1024 * 2);
    unsigned short* wq = wbf;                 // 768*256
    unsigned short* wp = wq + 768 * 256;      // 256*256
    unsigned short* w1 = wp + 256 * 256;      // 1024*256
    unsigned short* w2 = w1 + 1024 * 256;     // 256*1024

    // 0. one-time weight conversion to bf16
    cvt_bf16_kernel<<<768 * 256 / 1024, 256, 0, stream>>>(qkv_w, wq, 768 * 256);
    cvt_bf16_kernel<<<256 * 256 / 1024, 256, 0, stream>>>(proj_w, wp, 256 * 256);
    cvt_bf16_kernel<<<1024 * 256 / 1024, 256, 0, stream>>>(fc1w, w1, 1024 * 256);
    cvt_bf16_kernel<<<256 * 1024 / 1024, 256, 0, stream>>>(fc2w, w2, 256 * 1024);

    // 1. LN1 + shift + window partition -> bf16
    ln_kernel<<<MTOK / 8, 256, 0, stream>>>(x, n1g, n1b, xw, 1);

    // 2. QKV GEMM (100352 x 768 x 256) -> q (scaled), k, v
    gemm_bf16_kernel<<<dim3(MTOK / 128, 768 / 64), 256, 0, stream>>>(
        xw, wq, qkv_b, MTOK, 768, CC, 0, qb, kb, vb, nullptr, nullptr);

    // 3. windowed attention with rpb + shift + fg/bg saliency softmaxes
    attn_kernel<<<BWIN * HEADS, 128, 0, stream>>>(
        qb, kb, vb, rpb, amask, salfg, salbg, xw);

    // 4. proj GEMM + window-reverse + roll-back + residual -> x2 (fp32)
    gemm_bf16_kernel<<<dim3(MTOK / 128, CC / 64), 256, 0, stream>>>(
        xw, wp, proj_b, MTOK, CC, CC, 1, nullptr, nullptr, nullptr, x2, x);

    // 5. LN2 -> bf16 (reuse xw buffer)
    ln_kernel<<<MTOK / 8, 256, 0, stream>>>(x2, n2g, n2b, xw, 0);

    // 6. fc1 GEMM + exact GELU (100352 x 1024 x 256)
    gemm_bf16_kernel<<<dim3(MTOK / 128, 1024 / 64), 256, 0, stream>>>(
        xw, w1, fc1b, MTOK, 1024, CC, 2, fc1o, nullptr, nullptr, nullptr, nullptr);

    // 7. fc2 GEMM + residual (100352 x 256 x 1024) -> d_out
    gemm_bf16_kernel<<<dim3(MTOK / 128, CC / 64), 256, 0, stream>>>(
        fc1o, w2, fc2b, MTOK, CC, 1024, 3, nullptr, nullptr, nullptr, out, x2);
}
